// FlowLayer_84009560310172
// MI455X (gfx1250) — matrix operand order repaired and checked by reading, not compiled, not run on hardware
//
#include <hip/hip_runtime.h>
#include <hip/hip_bf16.h>

typedef __bf16 bf16;
typedef __attribute__((ext_vector_type(16))) __bf16 v16bf;
typedef __attribute__((ext_vector_type(8)))  __bf16 v8bf;
typedef __attribute__((ext_vector_type(8)))  float  v8f;
typedef unsigned int u32x4 __attribute__((ext_vector_type(4)));
typedef int          i32x4 __attribute__((ext_vector_type(4)));
typedef int          i32x8 __attribute__((ext_vector_type(8)));

#define B_     4
#define C_     256
#define H_     256
#define W_     256
#define HD_    64
#define WD_    64
#define CO_    256
#define ITERS_ 10

#if __has_builtin(__builtin_amdgcn_tensor_load_to_lds)
#define HAVE_TDM 1
#else
#define HAVE_TDM 0
#endif

// ---------------- workspace layout (bytes) ----------------
#define OFF_XN    ((size_t)0)                 // bf16 NHWC (B,H,W,C)        134217728
#define OFF_DWA   ((size_t)134217728)         // bf16 NHWC dwconv out       134217728
#define OFF_GRID1 ((size_t)268435456)         // f32 (B,HD,WD,2)               131072
#define OFF_GRID2 ((size_t)268566528)         //                               131072
#define OFF_ATT   ((size_t)268697600)         // f32 pooled NHWC             67108864
#define OFF_XSA   ((size_t)335806464)         // f32 pooled NHWC             67108864
#define OFF_XSB   ((size_t)402915328)         // f32 pooled NHWC             67108864
#define OFF_AGG1  ((size_t)470024192)         // f32 pooled NHWC             67108864
#define OFF_AGG2  ((size_t)537133056)         // f32 pooled NHWC             67108864
#define OFF_DWM   ((size_t)604241920)         // bf16 (B,HD,WD,512)          33554432
#define OFF_XM    ((size_t)637796352)         // f32 pooled NHWC             67108864
#define OFF_WTS   ((size_t)704905216)         // bf16 weights

// ---------------- TDM helper: 2-D bf16 tile, global -> LDS ----------------
// D# group0: count=1 | lds_addr | global_addr(57b) | type=2
// D# group1: data_size=1(2B) | tensor_dim0/1 | tile_dim0/1 | dim0 stride
#if HAVE_TDM
__device__ __forceinline__ void tdm_load_2d(unsigned lds_off,
                                            unsigned long long gaddr,
                                            unsigned tile_w, unsigned tile_h,
                                            unsigned long long stride_el) {
  u32x4 g0;
  g0[0] = 1u;                                             // count=1 (valid D#)
  g0[1] = lds_off;                                        // lds_addr (bytes)
  g0[2] = (unsigned)(gaddr & 0xffffffffull);              // global_addr[31:0]
  g0[3] = (unsigned)((gaddr >> 32) & 0x1ffffffull) | (2u << 30);  // ga[56:32] | type=2
  i32x8 g1;
  g1[0] = 0x00010000;                                     // data_size=1 -> 2 bytes
  g1[1] = (int)((tile_w & 0xffffu) << 16);                // tensor_dim0[15:0]
  g1[2] = (int)(((tile_w >> 16) & 0xffffu) | ((tile_h & 0xffffu) << 16)); // td0 hi | td1 lo
  g1[3] = (int)(((tile_h >> 16) & 0xffffu) | ((tile_w & 0xffffu) << 16)); // td1 hi | tile_dim0
  g1[4] = (int)(tile_h & 0xffffu);                        // tile_dim1 (tile_dim2=0)
  g1[5] = (int)(unsigned)(stride_el & 0xffffffffull);     // tensor_dim0_stride[31:0]
  g1[6] = (int)((stride_el >> 32) & 0xffffull);           // stride[47:32] | td1_stride lo=0
  g1[7] = 0;
  i32x4 z4 = {0, 0, 0, 0};
#if defined(__clang_major__) && __clang_major__ >= 23
  i32x8 z8 = {0, 0, 0, 0, 0, 0, 0, 0};
  __builtin_amdgcn_tensor_load_to_lds(g0, g1, z4, z4, z8, 0);
#else
  __builtin_amdgcn_tensor_load_to_lds(g0, g1, z4, z4, 0);
#endif
}
#endif

// ---------------- elementwise / prep kernels ----------------

__global__ void cvt_w_bf16(const float* __restrict__ src, bf16* __restrict__ dst, int n) {
  int i = blockIdx.x * blockDim.x + threadIdx.x;
  if (i < n) dst[i] = (bf16)src[i];
}

// BatchNorm (eval) NCHW f32 -> NHWC bf16
__global__ void bn_nhwc_bf16(const float* __restrict__ x,
                             const float* __restrict__ gamma,
                             const float* __restrict__ beta,
                             const float* __restrict__ mean,
                             const float* __restrict__ var,
                             bf16* __restrict__ xn) {
  long i = (long)blockIdx.x * blockDim.x + threadIdx.x;
  if (i >= (long)B_ * C_ * H_ * W_) return;
  int w = (int)(i % W_);
  int h = (int)((i / W_) % H_);
  int c = (int)((i / ((long)H_ * W_)) % C_);
  int b = (int)(i / ((long)C_ * H_ * W_));
  float s = rsqrtf(var[c] + 1e-5f);
  float y = (x[i] - mean[c]) * s * gamma[c] + beta[c];
  xn[(((long)(b * H_ + h) * W_ + w)) * C_ + c] = (bf16)y;
}

// grids: avgpool(grads)/Hd + base; grid1 = base+g, grid2 = base-g
__global__ void make_grids(const float* __restrict__ grads,
                           float* __restrict__ g1, float* __restrict__ g2) {
  int i = blockIdx.x * blockDim.x + threadIdx.x;   // (b,py,px)
  if (i >= B_ * HD_ * WD_) return;
  int px = i % WD_, py = (i / WD_) % HD_, b = i / (HD_ * WD_);
  float s0 = 0.f, s1 = 0.f;
  for (int dy = 0; dy < 4; ++dy)
    for (int dx = 0; dx < 4; ++dx) {
      int h = py * 4 + dy, w = px * 4 + dx;
      s0 += grads[(((long)b * 2 + 0) * H_ + h) * W_ + w];
      s1 += grads[(((long)b * 2 + 1) * H_ + h) * W_ + w];
    }
  float gx = s0 * (1.0f / 16.0f) * (1.0f / (float)HD_);
  float gy = s1 * (1.0f / 16.0f) * (1.0f / (float)HD_);
  float bx = -1.0f + 2.0f * px / (float)(WD_ - 1);
  float by = -1.0f + 2.0f * py / (float)(HD_ - 1);
  g1[2 * i + 0] = bx + gx;  g1[2 * i + 1] = by + gy;
  g2[2 * i + 0] = bx - gx;  g2[2 * i + 1] = by - gy;
}

// depthwise 3x3 SAME over full-res NHWC bf16 -> NHWC bf16
__global__ void dwconv_full(const bf16* __restrict__ xn,
                            const float* __restrict__ dw,
                            bf16* __restrict__ out) {
  int pix = blockIdx.x;          // b*H*W + h*W + w
  int c = threadIdx.x;           // 0..255
  int w = pix % W_, h = (pix / W_) % H_, b = pix / (H_ * W_);
  float acc = 0.f;
  #pragma unroll
  for (int dy = -1; dy <= 1; ++dy)
    #pragma unroll
    for (int dx = -1; dx <= 1; ++dx) {
      int hh = h + dy, ww = w + dx;
      if (hh >= 0 && hh < H_ && ww >= 0 && ww < W_) {
        float v = (float)xn[(((long)(b * H_ + hh) * W_ + ww)) * C_ + c];
        acc += v * dw[c * 9 + (dy + 1) * 3 + (dx + 1)];
      }
    }
  out[(long)pix * C_ + c] = (bf16)acc;
}

// ---------------- WMMA GEMM + fused 4x4 maxpool + activation ----------------
// A: full-res NHWC bf16 (M rows = the 16 pixels of one pooled cell)
// Wt: (256,256) bf16 row-major.  out: pooled NHWC f32.
// act==0: relu(max) ; act==1: sigmoid(max)   (both commute with maxpool)
__global__ __launch_bounds__(256)
void gemm_pool(const bf16* __restrict__ A, const bf16* __restrict__ Wt,
               float* __restrict__ out, int act) {
  __shared__ __align__(16) bf16 lds_a[16 * 256];
  int cell = blockIdx.x;                      // b*4096 + py*64 + px
  int px = cell % WD_, py = (cell / WD_) % HD_, b = cell / (HD_ * WD_);
  int t = threadIdx.x;
#if HAVE_TDM
  // TDM: 4 rows x 1024 bf16 (4 full-res pixels x 256 ch, row stride W*C)
  // lands in LDS exactly as lds_a[m*256 + k], m = 4*row + pixel-in-row.
  if (t < 32) {
    unsigned long long ga =
        (unsigned long long)(A + (((long)(b * H_ + py * 4) * W_ + px * 4)) * C_);
    tdm_load_2d((unsigned)(unsigned long long)&lds_a[0], ga,
                1024u, 4u, (unsigned long long)W_ * C_);
    __builtin_amdgcn_s_wait_tensorcnt(0);
  }
#else
  {   // manual stage 16x256 A panel
    int m = t & 15, kc = t >> 4;
    int h = py * 4 + (m >> 2), w = px * 4 + (m & 3);
    const int4* s = (const int4*)(A + (((long)(b * H_ + h) * W_ + w)) * C_ + kc * 16);
    int4* d = (int4*)(lds_a + m * 256 + kc * 16);
    d[0] = s[0]; d[1] = s[1];
  }
#endif
  __syncthreads();
  int lane = t & 31, wave = t >> 5;
  int mr = lane & 15, half = lane >> 4;
  #pragma unroll
  for (int nti = 0; nti < 2; ++nti) {
    int nt = wave * 2 + nti;
    const bf16* wrow = Wt + (long)(nt * 16 + (lane & 15)) * 256;
    __builtin_prefetch((const void*)wrow, 0, 0);     // global_prefetch_b8
    v8f acc = {};
    #pragma unroll
    for (int kt = 0; kt < 8; ++kt) {
      int k0 = kt * 32 + half * 16;
      v8bf lo = *(const v8bf*)(lds_a + mr * 256 + k0);
      v8bf hi = *(const v8bf*)(lds_a + mr * 256 + k0 + 8);
      v16bf a = __builtin_shufflevector(lo, hi,
                 0,1,2,3,4,5,6,7,8,9,10,11,12,13,14,15);
      v16bf bm = *(const v16bf*)(wrow + kt * 32 + half * 16);
      acc = __builtin_amdgcn_wmma_f32_16x16x32_bf16(false, a, false, bm,
                                                    (short)0, acc, false, false);
    }
    float mx = acc[0];
    #pragma unroll
    for (int j = 1; j < 8; ++j) mx = fmaxf(mx, acc[j]);
    mx = fmaxf(mx, __shfl_xor(mx, 16));       // combine M halves (wave32)
    if (lane < 16) {
      float v = act ? (1.0f / (1.0f + __expf(-mx))) : fmaxf(mx, 0.0f);
      out[(long)cell * CO_ + nt * 16 + lane] = v;
    }
  }
}

// ---------------- grid_sample * att ----------------
__global__ void sample_mul(const float* __restrict__ xc, const float* __restrict__ att,
                           const float* __restrict__ grid, float* __restrict__ xs) {
  int pix = blockIdx.x;                       // b*4096 + py*64 + px
  int c = threadIdx.x;                        // 0..255
  int b = pix / (HD_ * WD_);
  float gx = (grid[2 * pix + 0] + 1.0f) * 0.5f * (float)(WD_ - 1);
  float gy = (grid[2 * pix + 1] + 1.0f) * 0.5f * (float)(HD_ - 1);
  float x0 = floorf(gx), y0 = floorf(gy);
  float wx1 = gx - x0, wy1 = gy - y0;
  float acc = 0.f;
  #pragma unroll
  for (int j = 0; j < 4; ++j) {
    float xf = x0 + (float)(j & 1), yf = y0 + (float)(j >> 1);
    float wt = ((j & 1) ? wx1 : 1.f - wx1) * ((j >> 1) ? wy1 : 1.f - wy1);
    bool valid = (xf >= 0.f) && (xf <= (float)(WD_ - 1)) &&
                 (yf >= 0.f) && (yf <= (float)(HD_ - 1));
    int xi = (int)fminf(fmaxf(xf, 0.f), (float)(WD_ - 1));
    int yi = (int)fminf(fmaxf(yf, 0.f), (float)(HD_ - 1));
    float v = xc[(((long)(b * HD_ + yi) * WD_ + xi)) * CO_ + c];
    acc += valid ? wt * v : 0.f;
  }
  xs[(long)pix * CO_ + c] = acc * att[(long)pix * CO_ + c];
}

// ---------------- agg = add_w(256x512) @ [agg ; xs]  (in-place, WMMA) ----------------
__global__ __launch_bounds__(256)
void agg_update(float* __restrict__ agg, const float* __restrict__ xs,
                const bf16* __restrict__ Wadd) {
  __shared__ __align__(16) bf16 lds_a[16 * 512];
  int p0 = blockIdx.x * 16;
  int t = threadIdx.x;
  {   // stage 16x512 concat panel as bf16 (needs f32->bf16 convert, no TDM)
    int m = t & 15, kc = t >> 4;              // 32 f32 -> bf16 per thread
    const float* src = (kc < 8) ? (agg + (long)(p0 + m) * CO_ + kc * 32)
                                : (xs  + (long)(p0 + m) * CO_ + (kc - 8) * 32);
    bf16* d = lds_a + m * 512 + kc * 32;
    #pragma unroll
    for (int j = 0; j < 32; ++j) d[j] = (bf16)src[j];
  }
  __syncthreads();
  int lane = t & 31, wave = t >> 5;
  int mr = lane & 15, half = lane >> 4, nn = lane & 15;
  #pragma unroll
  for (int nti = 0; nti < 2; ++nti) {
    int nt = wave * 2 + nti;
    const bf16* wrow = Wadd + (long)(nt * 16 + nn) * 512;
    __builtin_prefetch((const void*)wrow, 0, 0);
    v8f acc = {};
    #pragma unroll
    for (int kt = 0; kt < 16; ++kt) {
      int k0 = kt * 32 + half * 16;
      v8bf lo = *(const v8bf*)(lds_a + mr * 512 + k0);
      v8bf hi = *(const v8bf*)(lds_a + mr * 512 + k0 + 8);
      v16bf a = __builtin_shufflevector(lo, hi,
                 0,1,2,3,4,5,6,7,8,9,10,11,12,13,14,15);
      v16bf bm = *(const v16bf*)(wrow + kt * 32 + half * 16);
      acc = __builtin_amdgcn_wmma_f32_16x16x32_bf16(false, a, false, bm,
                                                    (short)0, acc, false, false);
    }
    #pragma unroll
    for (int v = 0; v < 8; ++v) {             // D layout: M = v + 8*half, N = lane&15
      int mm = v + 8 * half;
      agg[(long)(p0 + mm) * CO_ + nt * 16 + nn] = acc[v];
    }
  }
}

// ---------------- merge depthwise (on concat(x1,x2)/ITERS), pooled res ----------------
__global__ void dwconv_merge(const float* __restrict__ x1, const float* __restrict__ x2,
                             const float* __restrict__ dw, bf16* __restrict__ out) {
  int pix = blockIdx.x;                       // b*4096 + py*64 + px
  int c = threadIdx.x;                        // 0..511
  int w = pix % WD_, h = (pix / WD_) % HD_, b = pix / (HD_ * WD_);
  const float* src = (c < CO_) ? x1 : x2;
  int cc = (c < CO_) ? c : c - CO_;
  float acc = 0.f;
  #pragma unroll
  for (int dy = -1; dy <= 1; ++dy)
    #pragma unroll
    for (int dx = -1; dx <= 1; ++dx) {
      int hh = h + dy, ww = w + dx;
      if (hh >= 0 && hh < HD_ && ww >= 0 && ww < WD_) {
        float v = src[(((long)(b * HD_ + hh) * WD_ + ww)) * CO_ + cc] * (1.0f / ITERS_);
        acc += v * dw[c * 9 + (dy + 1) * 3 + (dx + 1)];
      }
    }
  out[(long)pix * 512 + c] = (bf16)acc;
}

// ---------------- merge pointwise: xm = merge_pw(256x512) @ dwm ----------------
__global__ __launch_bounds__(256)
void gemm512(const bf16* __restrict__ A, const bf16* __restrict__ Wt,
             float* __restrict__ out) {
  __shared__ __align__(16) bf16 lds_a[16 * 512];
  int p0 = blockIdx.x * 16;
  int t = threadIdx.x;
#if HAVE_TDM
  if (t < 32) {   // TDM: 16 rows x 512 bf16, contiguous panel
    tdm_load_2d((unsigned)(unsigned long long)&lds_a[0],
                (unsigned long long)(A + (long)p0 * 512), 512u, 16u, 512ull);
    __builtin_amdgcn_s_wait_tensorcnt(0);
  }
#else
  {
    int m = t & 15, kc = t >> 4;              // 32 bf16 per thread
    const int4* s = (const int4*)(A + (long)(p0 + m) * 512 + kc * 32);
    int4* d = (int4*)(lds_a + m * 512 + kc * 32);
    d[0] = s[0]; d[1] = s[1]; d[2] = s[2]; d[3] = s[3];
  }
#endif
  __syncthreads();
  int lane = t & 31, wave = t >> 5;
  int mr = lane & 15, half = lane >> 4, nn = lane & 15;
  #pragma unroll
  for (int nti = 0; nti < 2; ++nti) {
    int nt = wave * 2 + nti;
    const bf16* wrow = Wt + (long)(nt * 16 + nn) * 512;
    __builtin_prefetch((const void*)wrow, 0, 0);
    v8f acc = {};
    #pragma unroll
    for (int kt = 0; kt < 16; ++kt) {
      int k0 = kt * 32 + half * 16;
      v8bf lo = *(const v8bf*)(lds_a + mr * 512 + k0);
      v8bf hi = *(const v8bf*)(lds_a + mr * 512 + k0 + 8);
      v16bf a = __builtin_shufflevector(lo, hi,
                 0,1,2,3,4,5,6,7,8,9,10,11,12,13,14,15);
      v16bf bm = *(const v16bf*)(wrow + kt * 32 + half * 16);
      acc = __builtin_amdgcn_wmma_f32_16x16x32_bf16(false, a, false, bm,
                                                    (short)0, acc, false, false);
    }
    #pragma unroll
    for (int v = 0; v < 8; ++v) {
      int mm = v + 8 * half;
      out[(long)(p0 + mm) * CO_ + nt * 16 + nn] = acc[v];
    }
  }
}

// ---------------- 4x bilinear upsample (align_corners) ----------------
__global__ void upsample4(const float* __restrict__ xm, float* __restrict__ out) {
  long i = (long)blockIdx.x * blockDim.x + threadIdx.x;   // NCHW output
  if (i >= (long)B_ * CO_ * H_ * W_) return;
  int wo = (int)(i % W_);
  int ho = (int)((i / W_) % H_);
  int c = (int)((i / ((long)H_ * W_)) % CO_);
  int b = (int)(i / ((long)CO_ * H_ * W_));
  float ph = ho * (float)(HD_ - 1) / (float)(H_ - 1);
  float pw = wo * (float)(WD_ - 1) / (float)(W_ - 1);
  int y0 = (int)ph, x0 = (int)pw;
  int y1 = min(y0 + 1, HD_ - 1), x1 = min(x0 + 1, WD_ - 1);
  float wy = ph - (float)y0, wx = pw - (float)x0;
  long base = (long)b * HD_;
  float v00 = xm[((base + y0) * WD_ + x0) * CO_ + c];
  float v01 = xm[((base + y0) * WD_ + x1) * CO_ + c];
  float v10 = xm[((base + y1) * WD_ + x0) * CO_ + c];
  float v11 = xm[((base + y1) * WD_ + x1) * CO_ + c];
  out[i] = (1.f - wy) * ((1.f - wx) * v00 + wx * v01)
         + wy * ((1.f - wx) * v10 + wx * v11);
}

// ---------------- host orchestration ----------------
extern "C" void kernel_launch(void* const* d_in, const int* in_sizes, int n_in,
                              void* d_out, int out_size, void* d_ws, size_t ws_size,
                              hipStream_t stream) {
  const float* x        = (const float*)d_in[0];
  const float* grads    = (const float*)d_in[1];
  const float* bn_gamma = (const float*)d_in[2];
  const float* bn_beta  = (const float*)d_in[3];
  const float* bn_mean  = (const float*)d_in[4];
  const float* bn_var   = (const float*)d_in[5];
  const float* f_conv_w[2] = { (const float*)d_in[6],  (const float*)d_in[10] };
  const float* f_att_dw[2] = { (const float*)d_in[7],  (const float*)d_in[11] };
  const float* f_att_pw[2] = { (const float*)d_in[8],  (const float*)d_in[12] };
  const float* f_add_w[2]  = { (const float*)d_in[9],  (const float*)d_in[13] };
  const float* merge_dw = (const float*)d_in[14];
  const float* merge_pw = (const float*)d_in[15];

  char* ws = (char*)d_ws;
  bf16*  xn    = (bf16*)(ws + OFF_XN);
  bf16*  dwA   = (bf16*)(ws + OFF_DWA);
  float* grid1 = (float*)(ws + OFF_GRID1);
  float* grid2 = (float*)(ws + OFF_GRID2);
  float* att   = (float*)(ws + OFF_ATT);
  float* xsA   = (float*)(ws + OFF_XSA);
  float* xsB   = (float*)(ws + OFF_XSB);
  float* agg1  = (float*)(ws + OFF_AGG1);
  float* agg2  = (float*)(ws + OFF_AGG2);
  bf16*  dwm   = (bf16*)(ws + OFF_DWM);
  float* xm    = (float*)(ws + OFF_XM);
  bf16*  wb    = (bf16*)(ws + OFF_WTS);
  bf16* w_conv[2] = { wb,            wb + 65536 };
  bf16* w_att[2]  = { wb + 131072,   wb + 196608 };
  bf16* w_add[2]  = { wb + 262144,   wb + 393216 };
  bf16* w_merge   = wb + 524288;

  // weight conversion to bf16
  cvt_w_bf16<<<256, 256, 0, stream>>>(f_conv_w[0], w_conv[0], 65536);
  cvt_w_bf16<<<256, 256, 0, stream>>>(f_conv_w[1], w_conv[1], 65536);
  cvt_w_bf16<<<256, 256, 0, stream>>>(f_att_pw[0], w_att[0], 65536);
  cvt_w_bf16<<<256, 256, 0, stream>>>(f_att_pw[1], w_att[1], 65536);
  cvt_w_bf16<<<512, 256, 0, stream>>>(f_add_w[0], w_add[0], 131072);
  cvt_w_bf16<<<512, 256, 0, stream>>>(f_add_w[1], w_add[1], 131072);
  cvt_w_bf16<<<512, 256, 0, stream>>>(merge_pw, w_merge, 131072);

  bn_nhwc_bf16<<<262144, 256, 0, stream>>>(x, bn_gamma, bn_beta, bn_mean, bn_var, xn);
  make_grids<<<64, 256, 0, stream>>>(grads, grid1, grid2);

  const int NCELL = B_ * HD_ * WD_;           // 16384
  for (int flow = 0; flow < 2; ++flow) {
    float* agg = flow ? agg2 : agg1;
    const float* grid = flow ? grid2 : grid1;
    dwconv_full<<<B_ * H_ * W_, 256, 0, stream>>>(xn, f_att_dw[flow], dwA);
    gemm_pool<<<NCELL, 256, 0, stream>>>(dwA, w_att[flow], att, 1);   // sigmoid
    gemm_pool<<<NCELL, 256, 0, stream>>>(xn, w_conv[flow], xsA, 0);   // relu
    hipMemcpyAsync(agg, xsA, (size_t)NCELL * CO_ * sizeof(float),
                   hipMemcpyDeviceToDevice, stream);
    float* cur = xsA; float* nxt = xsB;
    for (int it = 0; it < ITERS_; ++it) {
      sample_mul<<<NCELL, 256, 0, stream>>>(cur, att, grid, nxt);
      agg_update<<<NCELL / 16, 256, 0, stream>>>(agg, nxt, w_add[flow]);
      float* tmp = cur; cur = nxt; nxt = tmp;
    }
  }

  dwconv_merge<<<NCELL, 512, 0, stream>>>(agg1, agg2, merge_dw, dwm);
  gemm512<<<NCELL / 16, 256, 0, stream>>>(dwm, w_merge, xm);
  upsample4<<<262144, 256, 0, stream>>>(xm, (float*)d_out);
}
